// Block_12154757448460
// MI455X (gfx1250) — matrix operand order repaired and checked by reading, not compiled, not run on hardware
//
#include <hip/hip_runtime.h>
#include <hip/hip_bf16.h>

typedef _Float16 half_t;
typedef __attribute__((ext_vector_type(16))) _Float16 v16h;
typedef __attribute__((ext_vector_type(8)))  _Float16 v8h;
typedef __attribute__((ext_vector_type(8)))  float    v8f;

// ---------------------------------------------------------------------------
// WMMA helpers (CDNA5, wave32, 16x16x32 f16 -> f32)
// A-frag: lane = 16*hi + row; half i -> K = 16*hi + i (same order as B).
// A staged row-major in LDS, ld = K-tile.
// B-frag: lane = 16*hi + col; half i -> K = 16*hi + i.  B staged
// TRANSPOSED in LDS ([col][K]) so each lane reads 16 contiguous halves.
// ---------------------------------------------------------------------------
__device__ __forceinline__ v16h load_frag_a(const half_t* __restrict__ p0, int ld) {
    int lane = threadIdx.x & 31;
    const half_t* p = p0 + (lane & 15) * ld + (lane >> 4) * 16;
    v16h a;
#pragma unroll
    for (int i = 0; i < 8; ++i) a[i] = p[i];
#pragma unroll
    for (int i = 0; i < 8; ++i) a[i + 8] = p[8 + i];
    return a;
}

__device__ __forceinline__ v16h load_frag_bt(const half_t* __restrict__ p0, int ld) {
    int lane = threadIdx.x & 31;
    const half_t* p = p0 + (lane & 15) * ld + (lane >> 4) * 16;
    v16h b;
#pragma unroll
    for (int i = 0; i < 16; ++i) b[i] = p[i];
    return b;
}

__device__ __forceinline__ v8f wmma_f16(v16h a, v16h b, v8f c) {
    return __builtin_amdgcn_wmma_f32_16x16x32_f16(false, a, false, b,
                                                  (short)0, c, false, false);
}

// ---------------------------------------------------------------------------
// f32 -> f16 convert
// ---------------------------------------------------------------------------
__global__ void cvt_f16_kernel(const float* __restrict__ in,
                               half_t* __restrict__ out, int n) {
    int i = blockIdx.x * 256 + threadIdx.x;
    if (i < n) out[i] = (half_t)in[i];
}

// W f32 [K x N] row-major  ->  WT f16 [N x K] row-major (one-shot, tiny)
__global__ void cvt_w_transpose_kernel(const float* __restrict__ in,
                                       half_t* __restrict__ out, int K, int N) {
    int idx = blockIdx.x * 256 + threadIdx.x;
    if (idx >= K * N) return;
    int n = idx / K, k = idx - n * K;
    out[idx] = (half_t)in[(size_t)k * N + n];
}

// f16 [TB, N, C] -> f16 [TB, C, N] tiled transpose (for v spikes)
__global__ __launch_bounds__(256) void transpose_nc_kernel(
    const half_t* __restrict__ in, half_t* __restrict__ out, int Nn, int C) {
    __shared__ __align__(16) half_t tile[64][72];
    int nb = blockIdx.x, cb = blockIdx.y, tb = blockIdx.z;
    int tid = threadIdx.x;
#pragma unroll
    for (int c = 0; c < 2; ++c) {
        int cc = tid * 2 + c;                    // 0..511
        int row = cc >> 3, off = (cc & 7) * 8;   // row = n, off = c
        *(v8h*)&tile[row][off] =
            *(const v8h*)&in[((size_t)tb * Nn + nb * 64 + row) * C + cb * 64 + off];
    }
    __syncthreads();
#pragma unroll
    for (int c = 0; c < 2; ++c) {
        int cc = tid * 2 + c;
        int crow = cc >> 3, noff = (cc & 7) * 8; // crow = c, noff = n
        v8h vv;
#pragma unroll
        for (int e = 0; e < 8; ++e) vv[e] = tile[noff + e][crow];
        *(v8h*)&out[((size_t)tb * C + cb * 64 + crow) * Nn + nb * 64 + noff] = vv;
    }
}

// ---------------------------------------------------------------------------
// GEMM + BN epilogue:  out[m][n] = bn(A[m][:] @ W[:][n] + bias[n])
// A: f16 [M x K] row-major.  WT: f16 [N x K] row-major (pre-transposed).
// out: f32 [M x N].  Block tile 128x64, K-step 64, 8 waves, wave tile 32x32.
// ---------------------------------------------------------------------------
#define BM 128
#define BN 64
#define BK 64

__global__ __launch_bounds__(256) void gemm_bn_kernel(
    const half_t* __restrict__ A, const half_t* __restrict__ WT,
    const float* __restrict__ bias, const float* __restrict__ g,
    const float* __restrict__ beta, const float* __restrict__ rm,
    const float* __restrict__ rv, float* __restrict__ out,
    int M, int N, int K) {
    __shared__ __align__(16) half_t As[BM * BK];
    __shared__ __align__(16) half_t Bs[BN * BK];   // [n][k]

    int tid = threadIdx.x;
    int wave = tid >> 5, lane = tid & 31;
    int mi = wave >> 1, ni = wave & 1;
    int m0 = blockIdx.x * BM, n0 = blockIdx.y * BN;

    v8f acc[2][2] = {};

    for (int ko = 0; ko < K; ko += BK) {
        // stage A tile: 128x64 halves, 4 x 16B per thread (contiguous)
#pragma unroll
        for (int c = 0; c < 4; ++c) {
            int cc = tid * 4 + c;                 // 0..1023
            int row = cc >> 3, off = (cc & 7) * 8;
            *(v8h*)&As[row * BK + off] =
                *(const v8h*)&A[(size_t)(m0 + row) * K + ko + off];
        }
        // stage B tile from pre-transposed WT: 64x64 halves, 2 x 16B/thread
#pragma unroll
        for (int c = 0; c < 2; ++c) {
            int cc = tid * 2 + c;                 // 0..511
            int row = cc >> 3, off = (cc & 7) * 8;
            *(v8h*)&Bs[row * BK + off] =
                *(const v8h*)&WT[(size_t)(n0 + row) * K + ko + off];
        }
        // prefetch next K-tile while WMMAs run (global_prefetch_b8)
        if (ko + BK < K) {
            __builtin_prefetch(&A[(size_t)(m0 + (tid >> 1)) * K + ko + BK], 0, 3);
            if (tid < BN)
                __builtin_prefetch(&WT[(size_t)(n0 + tid) * K + ko + BK], 0, 3);
        }
        __syncthreads();

#pragma unroll
        for (int kk = 0; kk < 2; ++kk) {
            v16h a0 = load_frag_a(&As[(mi * 32) * BK + kk * 32], BK);
            v16h a1 = load_frag_a(&As[(mi * 32 + 16) * BK + kk * 32], BK);
            v16h b0 = load_frag_bt(&Bs[(ni * 32) * BK + kk * 32], BK);
            v16h b1 = load_frag_bt(&Bs[(ni * 32 + 16) * BK + kk * 32], BK);
            acc[0][0] = wmma_f16(a0, b0, acc[0][0]);
            acc[0][1] = wmma_f16(a0, b1, acc[0][1]);
            acc[1][0] = wmma_f16(a1, b0, acc[1][0]);
            acc[1][1] = wmma_f16(a1, b1, acc[1][1]);
        }
        __syncthreads();
    }

    // BN epilogue: y = (acc + bias - rm) * g*rsqrt(rv+eps) + beta
    int colL = lane & 15, hi = lane >> 4;
#pragma unroll
    for (int j = 0; j < 2; ++j) {
        int n = n0 + ni * 32 + j * 16 + colL;
        float s = g[n] * rsqrtf(rv[n] + 1e-5f);
        float o = (bias[n] - rm[n]) * s + beta[n];
#pragma unroll
        for (int i = 0; i < 2; ++i) {
#pragma unroll
            for (int r = 0; r < 8; ++r) {
                int m = m0 + mi * 32 + i * 16 + r + hi * 8;
                out[(size_t)m * N + n] = acc[i][j][r] * s + o;
            }
        }
    }
}

// ---------------------------------------------------------------------------
// Fused RetNet attention per (t*B+b, h):  o = ((q k^T) * scale * D) v
// q,k: f16 spikes [TB, N, C].  vt: f16 spikes pre-transposed [TB, C, N].
// o: f32 [TB, N, C].  64 q-rows per block, 16 key-blocks of 64, P in LDS.
// ---------------------------------------------------------------------------
__global__ __launch_bounds__(256) void attn_kernel(
    const half_t* __restrict__ q, const half_t* __restrict__ k,
    const half_t* __restrict__ vt, float* __restrict__ o,
    int Nn, int C, int Hh) {
    __shared__ __align__(16) half_t Qs[64 * 64];
    __shared__ __align__(16) half_t Ks[64 * 64];   // [key][d] == B^T for q.k^T
    __shared__ __align__(16) half_t VT[64 * 64];   // [d][key] == B^T for P.v
    __shared__ __align__(16) half_t Ps[64 * 64];   // [qrow][key]

    int tbh = blockIdx.x;
    int h = tbh % Hh, tb = tbh / Hh;
    int q0 = blockIdx.y * 64;
    int tid = threadIdx.x, wave = tid >> 5, lane = tid & 31;
    int mi = wave >> 1, ni = wave & 1;
    int colL = lane & 15, hi = lane >> 4;

    const half_t* qb  = q  + (size_t)tb * Nn * C + h * 64;
    const half_t* kb_ = k  + (size_t)tb * Nn * C + h * 64;
    const half_t* vtb = vt + ((size_t)tb * C + h * 64) * Nn;

    // stage Q tile (64x64)
#pragma unroll
    for (int c = 0; c < 2; ++c) {
        int cc = tid * 2 + c;
        int row = cc >> 3, off = (cc & 7) * 8;
        *(v8h*)&Qs[row * 64 + off] = *(const v8h*)&qb[(size_t)(q0 + row) * C + off];
    }

    float gamma = 1.0f - exp2f(-5.0f - (float)h);
    float lg = log2f(gamma);
    const float scl = 0.125f;                   // d^-0.5, d=64

    v8f accO[2] = {};

    for (int kb = 0; kb < Nn; kb += 64) {
        __syncthreads();                        // prev Ps/Ks/VT consumed
#pragma unroll
        for (int c = 0; c < 2; ++c) {
            int cc = tid * 2 + c;
            int row = cc >> 3, off = (cc & 7) * 8;
            *(v8h*)&Ks[row * 64 + off] =
                *(const v8h*)&kb_[(size_t)(kb + row) * C + off];
            *(v8h*)&VT[row * 64 + off] =
                *(const v8h*)&vtb[(size_t)row * Nn + kb + off];
        }
        if (kb + 64 < Nn) {                     // prefetch next key block
            __builtin_prefetch(&kb_[(size_t)(kb + 64 + (tid >> 2)) * C], 0, 3);
            __builtin_prefetch(&vtb[(size_t)(tid & 63) * Nn + kb + 64], 0, 3);
        }
        __syncthreads();

        // P tile: wave covers rows 16*mi, cols [32*ni, 32*ni+32)
        v16h aq0 = load_frag_a(&Qs[(mi * 16) * 64], 64);
        v16h aq1 = load_frag_a(&Qs[(mi * 16) * 64 + 32], 64);
#pragma unroll
        for (int j = 0; j < 2; ++j) {
            v8f p = {};
            p = wmma_f16(aq0, load_frag_bt(&Ks[(ni * 32 + j * 16) * 64], 64), p);
            p = wmma_f16(aq1, load_frag_bt(&Ks[(ni * 32 + j * 16) * 64 + 32], 64), p);
            int jc = kb + ni * 32 + j * 16 + colL;
#pragma unroll
            for (int r = 0; r < 8; ++r) {
                int iq = q0 + mi * 16 + r + hi * 8;
                float d = exp2f(lg * fabsf((float)(iq - jc)));
                Ps[(mi * 16 + r + hi * 8) * 64 + ni * 32 + j * 16 + colL] =
                    (half_t)(p[r] * scl * d);
            }
        }
        __syncthreads();

        // O += P @ V
#pragma unroll
        for (int kk = 0; kk < 2; ++kk) {
            v16h ap = load_frag_a(&Ps[(mi * 16) * 64 + kk * 32], 64);
#pragma unroll
            for (int j = 0; j < 2; ++j) {
                v16h bv = load_frag_bt(&VT[(ni * 32 + j * 16) * 64 + kk * 32], 64);
                accO[j] = wmma_f16(ap, bv, accO[j]);
            }
        }
    }

    float* ob = o + (size_t)tb * Nn * C + h * 64;
#pragma unroll
    for (int j = 0; j < 2; ++j)
#pragma unroll
        for (int r = 0; r < 8; ++r) {
            int row = q0 + mi * 16 + r + hi * 8;
            ob[(size_t)row * C + ni * 32 + j * 16 + colL] = accO[j][r];
        }
}

// ---------------------------------------------------------------------------
// LIF kernels (TAU=2, hard reset).  pre: f32 [4][S]; spikes out f16.
// ---------------------------------------------------------------------------
__global__ void lif_spike_kernel(const float* __restrict__ pre,
                                 half_t* __restrict__ sp, int S, float vth) {
    int i = blockIdx.x * 256 + threadIdx.x;
    if (i >= S) return;
    float v = 0.f;
#pragma unroll
    for (int t = 0; t < 4; ++t) {
        v = 0.5f * (v + pre[(size_t)t * S + i]);
        float s = (v >= vth) ? 1.f : 0.f;
        sp[(size_t)t * S + i] = (half_t)s;
        v *= (1.f - s);
    }
}

__global__ void lif_residual_kernel(const float* __restrict__ pre,
                                    const float* __restrict__ x,
                                    float* __restrict__ x1,
                                    half_t* __restrict__ x1h, int S) {
    int i = blockIdx.x * 256 + threadIdx.x;
    if (i >= S) return;
    float v = 0.f;
#pragma unroll
    for (int t = 0; t < 4; ++t) {
        v = 0.5f * (v + pre[(size_t)t * S + i]);
        float s = (v >= 1.f) ? 1.f : 0.f;
        v *= (1.f - s);
        float y = x[(size_t)t * S + i] + s;
        x1[(size_t)t * S + i] = y;
        x1h[(size_t)t * S + i] = (half_t)y;
    }
}

__global__ void lif_final_kernel(const float* __restrict__ pre,
                                 const float* __restrict__ x1,
                                 float* __restrict__ out, int S) {
    int i = blockIdx.x * 256 + threadIdx.x;
    if (i >= S) return;
    float v = 0.f;
#pragma unroll
    for (int t = 0; t < 4; ++t) {
        v = 0.5f * (v + pre[(size_t)t * S + i]);
        float s = (v >= 1.f) ? 1.f : 0.f;
        v *= (1.f - s);
        out[(size_t)t * S + i] = x1[(size_t)t * S + i] + s;
    }
}

// ---------------------------------------------------------------------------
extern "C" void kernel_launch(void* const* d_in, const int* in_sizes, int n_in,
                              void* d_out, int out_size, void* d_ws, size_t ws_size,
                              hipStream_t stream) {
    const int T = 4, B = 4, Nn = 1024, C = 512, HID = 2048, H = 8;
    const int TB = T * B;
    const int M = T * B * Nn;                  // 16384
    const int S1 = B * Nn * C;                 // 2,097,152
    const int S2 = B * Nn * HID;               // 8,388,608

    const float* x = (const float*)d_in[0];
    const float* Wq = (const float*)d_in[1],  *bq = (const float*)d_in[2];
    const float* gq = (const float*)d_in[3],  *betaq = (const float*)d_in[4];
    const float* rmq = (const float*)d_in[5], *rvq = (const float*)d_in[6];
    const float* Wk = (const float*)d_in[7],  *bk = (const float*)d_in[8];
    const float* gk = (const float*)d_in[9],  *betak = (const float*)d_in[10];
    const float* rmk = (const float*)d_in[11],*rvk = (const float*)d_in[12];
    const float* Wv = (const float*)d_in[13], *bv = (const float*)d_in[14];
    const float* gv = (const float*)d_in[15], *betav = (const float*)d_in[16];
    const float* rmv = (const float*)d_in[17],*rvv = (const float*)d_in[18];
    const float* Wp = (const float*)d_in[19], *bp = (const float*)d_in[20];
    const float* gp = (const float*)d_in[21], *betap = (const float*)d_in[22];
    const float* rmp = (const float*)d_in[23],*rvp = (const float*)d_in[24];
    const float* W1 = (const float*)d_in[25], *b1 = (const float*)d_in[26];
    const float* g1 = (const float*)d_in[27], *beta1 = (const float*)d_in[28];
    const float* rm1 = (const float*)d_in[29],*rv1 = (const float*)d_in[30];
    const float* W2 = (const float*)d_in[31], *b2 = (const float*)d_in[32];
    const float* g2 = (const float*)d_in[33], *beta2 = (const float*)d_in[34];
    const float* rm2 = (const float*)d_in[35],*rv2 = (const float*)d_in[36];

    // workspace carving
    char* w = (char*)d_ws;
    size_t off = 0;
    auto carve = [&](size_t bytes) -> void* {
        void* p = w + off;
        off = (off + bytes + 255) & ~(size_t)255;
        return p;
    };
    half_t* XH  = (half_t*)carve((size_t)M * C * 2);
    half_t* WQT = (half_t*)carve((size_t)C * C * 2);   // pre-transposed [N x K]
    half_t* WKT = (half_t*)carve((size_t)C * C * 2);
    half_t* WVT = (half_t*)carve((size_t)C * C * 2);
    half_t* WPT = (half_t*)carve((size_t)C * C * 2);
    half_t* W1T = (half_t*)carve((size_t)C * HID * 2);
    half_t* W2T = (half_t*)carve((size_t)HID * C * 2);
    half_t* SQ  = (half_t*)carve((size_t)M * C * 2);
    half_t* SK  = (half_t*)carve((size_t)M * C * 2);
    half_t* SV  = (half_t*)carve((size_t)M * C * 2);
    half_t* SVT = (half_t*)carve((size_t)M * C * 2);   // [TB, C, N]
    float*  PRE = (float*) carve((size_t)M * HID * 4); // shared pre-act scratch
    float*  O   = (float*) carve((size_t)M * C * 4);
    half_t* R   = (half_t*)carve((size_t)M * C * 2);
    float*  X1F = (float*) carve((size_t)M * C * 4);
    half_t* X1H = (half_t*)carve((size_t)M * C * 2);
    half_t* HH  = (half_t*)carve((size_t)M * HID * 2);

    cvt_f16_kernel<<<(M * C + 255) / 256, 256, 0, stream>>>(x, XH, M * C);
    auto cvtT = [&](const float* src, half_t* dst, int K, int N) {
        cvt_w_transpose_kernel<<<(K * N + 255) / 256, 256, 0, stream>>>(src, dst, K, N);
    };
    cvtT(Wq, WQT, C, C); cvtT(Wk, WKT, C, C);
    cvtT(Wv, WVT, C, C); cvtT(Wp, WPT, C, C);
    cvtT(W1, W1T, C, HID); cvtT(W2, W2T, HID, C);

    auto gemm = [&](const half_t* A, const half_t* Wt, const float* bias,
                    const float* g, const float* beta, const float* rm,
                    const float* rv, float* out, int N, int K) {
        dim3 grid(M / BM, N / BN);
        gemm_bn_kernel<<<grid, 256, 0, stream>>>(A, Wt, bias, g, beta, rm, rv,
                                                 out, M, N, K);
    };

    // q, k, v projections + LIF
    gemm(XH, WQT, bq, gq, betaq, rmq, rvq, PRE, C, C);
    lif_spike_kernel<<<(S1 + 255) / 256, 256, 0, stream>>>(PRE, SQ, S1, 1.f);
    gemm(XH, WKT, bk, gk, betak, rmk, rvk, PRE, C, C);
    lif_spike_kernel<<<(S1 + 255) / 256, 256, 0, stream>>>(PRE, SK, S1, 1.f);
    gemm(XH, WVT, bv, gv, betav, rmv, rvv, PRE, C, C);
    lif_spike_kernel<<<(S1 + 255) / 256, 256, 0, stream>>>(PRE, SV, S1, 1.f);

    // pre-transpose v spikes: [TB,N,C] -> [TB,C,N]
    transpose_nc_kernel<<<dim3(Nn / 64, C / 64, TB), 256, 0, stream>>>(SV, SVT,
                                                                       Nn, C);
    // fused attention
    attn_kernel<<<dim3(TB * H, Nn / 64), 256, 0, stream>>>(SQ, SK, SVT, O,
                                                           Nn, C, H);
    // retention LIF (v_th = 0.5)
    lif_spike_kernel<<<(S1 + 255) / 256, 256, 0, stream>>>(O, R, S1, 0.5f);

    // projection + LIF + residual 1
    gemm(R, WPT, bp, gp, betap, rmp, rvp, PRE, C, C);
    lif_residual_kernel<<<(S1 + 255) / 256, 256, 0, stream>>>(PRE, x, X1F, X1H, S1);

    // MLP
    gemm(X1H, W1T, b1, g1, beta1, rm1, rv1, PRE, HID, C);
    lif_spike_kernel<<<(S2 + 255) / 256, 256, 0, stream>>>(PRE, HH, S2, 1.f);
    gemm(HH, W2T, b2, g2, beta2, rm2, rv2, PRE, C, HID);
    lif_final_kernel<<<(S1 + 255) / 256, 256, 0, stream>>>(PRE, X1F,
                                                           (float*)d_out, S1);
}
